// SelfAttention_51934744543351
// MI455X (gfx1250) — compile-verified, matrix operand order checked by reading, not run on hardware
//
#include <hip/hip_runtime.h>
#include <hip/hip_bf16.h>
#include <stdint.h>

// ---------------------------------------------------------------------------
// SAGAN self-attention, CDNA5 (gfx1250) WMMA bf16, flash-style streaming.
// B=4, N=4096, C=64, d=8.
// Phase 1 (WMMA): x tile = A, W^T staged bf16 in LDS = B; 10 WMMAs/wave.
// Phase 2 (WMMA): streaming attention, 4 waves/block, 16 rows/wave, key tiles
//   of 32; unshifted exact softmax (logits O(1), softmax shift-invariant):
//   row sums accumulate per lane, one cross-lane reduction at the end.
// V^T tiles are double-buffered in LDS via GLOBAL_LOAD_ASYNC_TO_LDS_B128
// (ASYNCcnt): tile i+1's DMA overlaps tile i's WMMAs.
//
// Fragment layout (CDNA5 ISA 7.12.2, wave32):
//   A 16x32 bf16 : lane=16h+m row m; VGPR v<4 -> K=2v+8h,2v+1+8h; v>=4 -> +16.
//   B 32x16 bf16 : lane=16h+n col n; same per-lane K pattern.
//   C/D 16x16 f32: lane=16h+n col n; VGPR v -> row v+8h.
// ---------------------------------------------------------------------------

typedef __attribute__((ext_vector_type(16))) __bf16 v16bf;
typedef __attribute__((ext_vector_type(8)))  float  v8f;
typedef __attribute__((ext_vector_type(4)))  int    v4i;
typedef __attribute__((ext_vector_type(2)))  __bf16 v2bf;

union BF16x16 { unsigned int u[8]; v16bf v; };

#if defined(__gfx1250__) && \
    __has_builtin(__builtin_amdgcn_global_load_async_to_lds_b128) && \
    __has_builtin(__builtin_amdgcn_s_wait_asynccnt)
#define USE_ASYNC_LDS 1
#else
#define USE_ASYNC_LDS 0
#endif

#define GPTR(p) ((__attribute__((address_space(1))) v4i*)(v4i*)(p))
#define LPTR(p) ((__attribute__((address_space(3))) v4i*)(v4i*)(p))

// native bf16 converts (v_cvt_pk_bf16_f32 on gfx1250; RNE)
__device__ __forceinline__ unsigned short f2bf(float f) {
    return __builtin_bit_cast(unsigned short, (__bf16)f);
}
__device__ __forceinline__ unsigned int pack_bf2(float a, float b) {
#if __has_builtin(__builtin_amdgcn_cvt_pk_bf16_f32)
    return __builtin_bit_cast(unsigned int, __builtin_amdgcn_cvt_pk_bf16_f32(a, b));
#else
    v2bf t; t[0] = (__bf16)a; t[1] = (__bf16)b;
    return __builtin_bit_cast(unsigned int, t);
#endif
}

#define NB   4
#define NTOK 4096
#define CDIM 64
#define DDIM 8

// ---------------------------------------------------------------------------
// Phase 1: QKV projection via WMMA. 128 threads = 4 waves = 64 tokens/block.
// grid = B*N/64 = 256 blocks.
// ---------------------------------------------------------------------------
__global__ __launch_bounds__(128) void qkv_kernel(
    const float* __restrict__ x,
    const float* __restrict__ Wq, const float* __restrict__ bq,
    const float* __restrict__ Wk, const float* __restrict__ bk,
    const float* __restrict__ Wv, const float* __restrict__ bv,
    unsigned short* __restrict__ Qb,    // [B][N][8] bf16 row-major
    unsigned short* __restrict__ Kb,    // [B][N][8] bf16 row-major
    unsigned short* __restrict__ Vt)    // [B][C][N] bf16 (V transposed)
{
    __shared__ __align__(4) unsigned short wt[80][72];  // W^T bf16, rows padded
    __shared__ float sbias[80];

    const int tid = threadIdx.x;
    for (int i = tid; i < CDIM * CDIM; i += 128) {
        const int c = i >> 6, j = i & 63;
        wt[j][c] = f2bf(Wv[i]);
    }
    for (int i = tid; i < CDIM * DDIM; i += 128) {
        const int c = i >> 3, j = i & 7;
        wt[64 + j][c] = f2bf(Wq[i]);
        wt[72 + j][c] = f2bf(Wk[i]);
    }
    if (tid < 80) sbias[tid] = (tid < 64) ? bv[tid] : (tid < 72 ? bq[tid - 64] : bk[tid - 72]);
    __syncthreads();

    const int wave = tid >> 5, lane = tid & 31;
    const int m = lane & 15, h = lane >> 4;
    const int b   = blockIdx.x >> 6;
    const int nb0 = ((blockIdx.x & 63) << 6) + wave * 16;   // token base in batch
    const int token = b * NTOK + nb0 + m;

    // A fragments from global x (f32 -> bf16 pairs), chunk0 c:[0,32) chunk1 c:[32,64)
    const float* xr = x + (size_t)token * CDIM;
    BF16x16 a0, a1;
#pragma unroll
    for (int v = 0; v < 4; ++v) {
        float2 t0 = *(const float2*)(xr + 2 * v + 8 * h);
        float2 t1 = *(const float2*)(xr + 16 + 2 * v + 8 * h);
        float2 t2 = *(const float2*)(xr + 32 + 2 * v + 8 * h);
        float2 t3 = *(const float2*)(xr + 48 + 2 * v + 8 * h);
        a0.u[v]     = pack_bf2(t0.x, t0.y);
        a0.u[v + 4] = pack_bf2(t1.x, t1.y);
        a1.u[v]     = pack_bf2(t2.x, t2.y);
        a1.u[v + 4] = pack_bf2(t3.x, t3.y);
    }

#pragma unroll
    for (int t = 0; t < 5; ++t) {
        const int j = t * 16 + m;                 // output column for this lane
        BF16x16 b0, b1;
#pragma unroll
        for (int v = 0; v < 4; ++v) {
            b0.u[v]     = *(const unsigned int*)&wt[j][2 * v + 8 * h];
            b0.u[v + 4] = *(const unsigned int*)&wt[j][16 + 2 * v + 8 * h];
            b1.u[v]     = *(const unsigned int*)&wt[j][32 + 2 * v + 8 * h];
            b1.u[v + 4] = *(const unsigned int*)&wt[j][48 + 2 * v + 8 * h];
        }
        const float bias = sbias[j];
        v8f c = {bias, bias, bias, bias, bias, bias, bias, bias};
        c = __builtin_amdgcn_wmma_f32_16x16x32_bf16(false, a0.v, false, b0.v,
                                                    (short)0, c, false, false);
        c = __builtin_amdgcn_wmma_f32_16x16x32_bf16(false, a1.v, false, b1.v,
                                                    (short)0, c, false, false);

        if (t < 4) {
            // V^T: column = 16t+m, rows = tokens nb0+v+8h; token pairs pack to u32
            unsigned short* vcol = Vt + ((size_t)(b * CDIM + t * 16 + m)) * NTOK + nb0 + 8 * h;
#pragma unroll
            for (int i = 0; i < 4; ++i)
                *(unsigned int*)(vcol + 2 * i) = pack_bf2(c[2 * i], c[2 * i + 1]);
        } else {
            // combined q|k tile: cols 0..7 -> q, 8..15 -> k (row-major scatter)
#pragma unroll
            for (int v = 0; v < 8; ++v) {
                const size_t tok = (size_t)(b * NTOK + nb0 + v + 8 * h);
                if (m < 8) Qb[tok * DDIM + m]       = f2bf(c[v]);
                else       Kb[tok * DDIM + (m - 8)] = f2bf(c[v]);
            }
        }
    }
}

// ---------------------------------------------------------------------------
// Phase 2: streaming attention with WMMA, double-buffered async V staging.
// grid = B*(N/64) = 256 blocks; 128 threads = 4 waves; 16 query rows/wave.
// ---------------------------------------------------------------------------
__global__ __launch_bounds__(128) void attn_kernel(
    const float* __restrict__ x,
    const unsigned short* __restrict__ Qb,
    const unsigned short* __restrict__ Kb,
    const unsigned short* __restrict__ Vt,
    const float* __restrict__ gamma,
    float* __restrict__ out)
{
    __shared__ __align__(16) unsigned short vt_lds[2][CDIM][40]; // V^T tiles, 2 buffers
    __shared__ __align__(16) unsigned short p_lds[4][16][40];    // per-wave P

    const int tid  = threadIdx.x;
    const int wave = tid >> 5;
    const int lane = tid & 31;
    const int m    = lane & 15;
    const int h    = lane >> 4;

    const int b     = blockIdx.x >> 6;
    const int q0    = (blockIdx.x & 63) << 6;
    const int qbase = q0 + wave * 16;

    // Q A-fragment: only K=0..7 (lanes 0-15, VGPRs 0-3) non-zero (d=8 pad to 32)
    BF16x16 aq;
#pragma unroll
    for (int i = 0; i < 8; ++i) aq.u[i] = 0u;
    if (h == 0) {
        const unsigned int* qr =
            (const unsigned int*)(Qb + ((size_t)(b * NTOK + qbase + m)) * DDIM);
#pragma unroll
        for (int v = 0; v < 4; ++v) aq.u[v] = qr[v];
    }

    v8f acc[4];
#pragma unroll
    for (int t = 0; t < 4; ++t) acc[t] = (v8f){0.f,0.f,0.f,0.f,0.f,0.f,0.f,0.f};
    float lsum[8];
#pragma unroll
    for (int v = 0; v < 8; ++v) lsum[v] = 0.f;

    const v8f zeroC = (v8f){0.f,0.f,0.f,0.f,0.f,0.f,0.f,0.f};

    // per-thread staging address: thread -> (column c, 16-key half)
    const int sc    = tid >> 1;
    const int shalf = tid & 1;
    const unsigned short* vsrc0 =
        Vt + ((size_t)(b * CDIM + sc)) * NTOK + shalf * 16;     // + k0 each iter

#if USE_ASYNC_LDS
    // prologue: DMA tile 0 into buffer 0
    __builtin_amdgcn_global_load_async_to_lds_b128(
        GPTR(vsrc0), LPTR(&vt_lds[0][sc][shalf * 16]), 0, 0);
    __builtin_amdgcn_global_load_async_to_lds_b128(
        GPTR(vsrc0), LPTR(&vt_lds[0][sc][shalf * 16]), 16, 0);
#endif

    for (int k0 = 0; k0 < NTOK; k0 += 32) {
#if USE_ASYNC_LDS
        const int cur = (k0 >> 5) & 1;
        if (k0 + 32 < NTOK) {
            // DMA next tile into the other buffer; overlaps this tile's compute
            const unsigned short* vnext = vsrc0 + k0 + 32;
            __builtin_amdgcn_global_load_async_to_lds_b128(
                GPTR(vnext), LPTR(&vt_lds[cur ^ 1][sc][shalf * 16]), 0, 0);
            __builtin_amdgcn_global_load_async_to_lds_b128(
                GPTR(vnext), LPTR(&vt_lds[cur ^ 1][sc][shalf * 16]), 16, 0);
            __builtin_amdgcn_s_wait_asynccnt(2);   // oldest 2 (= this tile) done
        } else {
            __builtin_amdgcn_s_wait_asynccnt(0);
        }
#else
        const int cur = 0;
        {
            const uint4* src = (const uint4*)(vsrc0 + k0);
            uint4 d0 = src[0], d1 = src[1];
            uint4* dst = (uint4*)&vt_lds[0][sc][shalf * 16];
            dst[0] = d0; dst[1] = d1;
        }
#endif
        // prefetch the streaming dimension a couple of tiles ahead
        if (k0 + 64 < NTOK) {
            __builtin_prefetch(vsrc0 + k0 + 64, 0, 0);
            __builtin_prefetch(Kb + ((size_t)(b * NTOK + k0 + 64)) * DDIM + (tid & 63) * DDIM, 0, 0);
        }

        // ---- K^T B-fragments for two 16-key halves ----
        BF16x16 kf0, kf1;
#pragma unroll
        for (int i = 0; i < 8; ++i) { kf0.u[i] = 0u; kf1.u[i] = 0u; }
        if (h == 0) {
            const unsigned int* kr0 =
                (const unsigned int*)(Kb + ((size_t)(b * NTOK + k0 + m)) * DDIM);
            const unsigned int* kr1 = kr0 + 16 * 4;
#pragma unroll
            for (int v = 0; v < 4; ++v) { kf0.u[v] = kr0[v]; kf1.u[v] = kr1[v]; }
        }
        __syncthreads();                        // all waves' staging visible

        // ---- S = Q K^T ----
        v8f s0 = __builtin_amdgcn_wmma_f32_16x16x32_bf16(false, aq.v, false, kf0.v,
                                                         (short)0, zeroC, false, false);
        v8f s1 = __builtin_amdgcn_wmma_f32_16x16x32_bf16(false, aq.v, false, kf1.v,
                                                         (short)0, zeroC, false, false);

        // ---- P = exp(S) (exact, unshifted); per-lane row sums ----
#pragma unroll
        for (int v = 0; v < 8; ++v) {
            const float p0 = __expf(s0[v]);
            const float p1 = __expf(s1[v]);
            lsum[v] += p0 + p1;
            const unsigned int pp = pack_bf2(p0, p1);   // one packed cvt
            const int r = v + 8 * h;
            p_lds[wave][r][m]      = (unsigned short)pp;          // ds_store_b16
            p_lds[wave][r][m + 16] = (unsigned short)(pp >> 16);  // ds_store_b16_d16_hi
        }
        // intra-wave LDS is in-order; only compiler-level ordering needed
        asm volatile("" ::: "memory");

        BF16x16 pa;
#pragma unroll
        for (int v = 0; v < 4; ++v) {
            pa.u[v]     = *(const unsigned int*)&p_lds[wave][m][2 * v + 8 * h];
            pa.u[v + 4] = *(const unsigned int*)&p_lds[wave][m][16 + 2 * v + 8 * h];
        }

        // ---- O += P @ V ----
#pragma unroll
        for (int t = 0; t < 4; ++t) {
            BF16x16 vf;
            const int c = t * 16 + m;
#pragma unroll
            for (int v = 0; v < 4; ++v) {
                vf.u[v]     = *(const unsigned int*)&vt_lds[cur][c][2 * v + 8 * h];
                vf.u[v + 4] = *(const unsigned int*)&vt_lds[cur][c][16 + 2 * v + 8 * h];
            }
            acc[t] = __builtin_amdgcn_wmma_f32_16x16x32_bf16(false, pa.v, false, vf.v,
                                                             (short)0, acc[t], false, false);
        }
        __syncthreads();   // all waves done with vt_lds[cur] before it's re-filled
    }

    // ---- one-time row-sum reduction across the 16-lane halves ----
#pragma unroll
    for (int v = 0; v < 8; ++v) {
#pragma unroll
        for (int off = 1; off < 16; off <<= 1)
            lsum[v] += __shfl_xor(lsum[v], off, 32);
    }

    // ---- epilogue: out = gamma * (acc / l) + x ----
    const float g = gamma[0];
    float rinv[8];
#pragma unroll
    for (int v = 0; v < 8; ++v) rinv[v] = 1.0f / lsum[v];
#pragma unroll
    for (int t = 0; t < 4; ++t) {
#pragma unroll
        for (int v = 0; v < 8; ++v) {
            const int row = qbase + v + 8 * h;
            const int col = t * 16 + m;
            const size_t idx = ((size_t)(b * NTOK + row)) * CDIM + col;
            out[idx] = g * acc[t][v] * rinv[v] + x[idx];
        }
    }
}

// ---------------------------------------------------------------------------
extern "C" void kernel_launch(void* const* d_in, const int* in_sizes, int n_in,
                              void* d_out, int out_size, void* d_ws, size_t ws_size,
                              hipStream_t stream) {
    const float* x     = (const float*)d_in[0];
    const float* Wq    = (const float*)d_in[1];
    const float* bq    = (const float*)d_in[2];
    const float* Wk    = (const float*)d_in[3];
    const float* bk    = (const float*)d_in[4];
    const float* Wv    = (const float*)d_in[5];
    const float* bv    = (const float*)d_in[6];
    const float* gamma = (const float*)d_in[7];
    float* out = (float*)d_out;

    // workspace: Qb (256KB) | Kb (256KB) | Vt (2MB), all bf16
    unsigned short* Qb = (unsigned short*)d_ws;
    unsigned short* Kb = Qb + (size_t)NB * NTOK * DDIM;
    unsigned short* Vt = Kb + (size_t)NB * NTOK * DDIM;

    qkv_kernel<<<dim3(256), dim3(128), 0, stream>>>(x, Wq, bq, Wk, bk, Wv, bv, Qb, Kb, Vt);
    attn_kernel<<<dim3(256), dim3(128), 0, stream>>>(x, Qb, Kb, Vt, gamma, out);
}
